// AdaptiveDirectionGraphConvLayer_54065048322062
// MI455X (gfx1250) — matrix operand order repaired and checked by reading, not compiled, not run on hardware
//
#include <hip/hip_runtime.h>
#include <hip/hip_bf16.h>
#include <math.h>

typedef __attribute__((ext_vector_type(16))) __bf16 v16bf;
typedef __attribute__((ext_vector_type(8)))  __bf16 v8bf;
typedef __attribute__((ext_vector_type(8)))  float  v8f;

#define N_PTS 2048
#define KK    14
#define CC    512
#define OUTC  512
#define NE    78            // (K-1)(K-2)/2 edges among nodes 0..12
#define MROWS (N_PTS*KK)    // 28672

// ---------------- workspace layout (bytes) ----------------
// [0,2048)          sum[512]   f32
// [2048,4096)       sumsq[512] f32
// [4096,6144)       u[512]     f32
// [6144,6208)       bconst    f32
// [8192,+512KB)     W_merged bf16
// [+512KB,+1MB)     W_orig   bf16
// [8192+1MB, +28MB) agg  bf16 (MROWS x C)
// [.., +28MB)       x    bf16 (MROWS x C)
#define WS_WM   (8192)
#define WS_WO   (8192 + 524288)
#define WS_AGG  (8192 + 1048576)
#define WS_XB   (8192 + 1048576 + (size_t)MROWS*CC*2)

// ---- kernel 0: weight f32->bf16 conversion + zero stat accumulators ----
__global__ void prep_kernel(const float* __restrict__ Wm, const float* __restrict__ Wo,
                            __bf16* __restrict__ wm_b, __bf16* __restrict__ wo_b,
                            float* __restrict__ sum, float* __restrict__ sumsq) {
  int idx = blockIdx.x * blockDim.x + threadIdx.x;
  if (idx < CC) { sum[idx] = 0.f; sumsq[idx] = 0.f; }
  if (idx < OUTC * CC) {
    wm_b[idx] = (__bf16)Wm[idx];
    wo_b[idx] = (__bf16)Wo[idx];
  }
}

// ---- kernel 1: per-channel sum / sumsq of gap over (n, edges) ----
__global__ void stats_kernel(const float* __restrict__ x,
                             float* __restrict__ sum, float* __restrict__ sumsq) {
  int c = blockIdx.x * 256 + threadIdx.x;   // gridDim.x = 2
  int nbase = blockIdx.y * 64;              // gridDim.y = 32
  float acc = 0.f, accq = 0.f;
  for (int nn = 0; nn < 64; ++nn) {
    int n = nbase + nn;
    const float* xp = x + (size_t)n * KK * CC + c;
    float xv[KK];
    #pragma unroll
    for (int m = 0; m < KK; ++m) xv[m] = xp[m * CC];
    float d[13];
    #pragma unroll
    for (int m = 0; m < 13; ++m) d[m] = fabsf(xv[m] - xv[13]);
    #pragma unroll
    for (int i = 0; i < 12; ++i)
      for (int j = i + 1; j < 13; ++j) {
        float g = 15.f * (d[i] - d[j]);
        acc += g;
        accq += g * g;
      }
  }
  atomicAdd(&sum[c], acc);
  atomicAdd(&sumsq[c], accq);
}

// ---- kernel 2: fold BN into u[c] and scalar bconst ----
__global__ void finalize_kernel(const float* __restrict__ gamma, const float* __restrict__ beta,
                                const float* __restrict__ wdir,
                                const float* __restrict__ sum, const float* __restrict__ sumsq,
                                float* __restrict__ u, float* __restrict__ bconst) {
  __shared__ float red[256];
  int t = threadIdx.x;
  const float inv_cnt = 1.0f / ((float)N_PTS * (float)NE);
  float part = 0.f;
  for (int c = t; c < CC; c += 256) {
    float mean = sum[c] * inv_cnt;
    float var  = sumsq[c] * inv_cnt - mean * mean;
    float gp   = gamma[c] * rsqrtf(var + 1e-5f);
    u[c] = gp * wdir[c];
    part += (beta[c] - mean * gp) * wdir[c];
  }
  red[t] = part;
  __syncthreads();
  for (int s = 128; s > 0; s >>= 1) {
    if (t < s) red[t] += red[t + s];
    __syncthreads();
  }
  if (t == 0) *bconst = red[0];
}

// ---- kernel 3: per-n edge gates, adjacency normalize, agg + x (bf16 out) ----
__global__ void adj_agg_kernel(const float* __restrict__ x, const float* __restrict__ adj,
                               const int* __restrict__ ei, const int* __restrict__ ej,
                               const float* __restrict__ u, const float* __restrict__ bconst,
                               __bf16* __restrict__ agg_b, __bf16* __restrict__ xb) {
  __shared__ float xs[KK * CC];   // 28 KB
  __shared__ float us[CC];        // 2 KB
  __shared__ float adjw[KK * KK];
  int n = blockIdx.x;
  int t = threadIdx.x;
  const float* xn = x + (size_t)n * KK * CC;
  for (int idx = t; idx < KK * CC; idx += 256) xs[idx] = xn[idx];
  for (int idx = t; idx < CC; idx += 256) us[idx] = u[idx];
  if (t < KK * KK) adjw[t] = adj[t];
  __syncthreads();
  if (t < NE) {
    int i = ei[t], j = ej[t];
    const float* xi = xs + i * CC;
    const float* xj = xs + j * CC;
    const float* xg = xs + 13 * CC;
    float acc = 0.f;
    for (int c = 0; c < CC; ++c) {
      float g = 15.f * (fabsf(xi[c] - xg[c]) - fabsf(xj[c] - xg[c]));
      acc += g * us[c];
    }
    float s = 4.0f * (acc + *bconst);
    float a = 1.f / (1.f + expf(-s));
    adjw[i * KK + j] = 2.f * a * adj[i * KK + j];
    adjw[j * KK + i] = 2.f * (1.f - a) * adj[j * KK + i];
  }
  __syncthreads();
  if (t < KK) {
    float norm = 0.f;
    for (int m = 0; m < KK; ++m) norm += fabsf(adjw[t * KK + m]);
    float inv = 1.f / fmaxf(norm, 1e-12f);
    for (int m = 0; m < KK; ++m) adjw[t * KK + m] *= inv;
  }
  __syncthreads();
  __bf16* outp = agg_b + (size_t)n * KK * CC;
  __bf16* xout = xb    + (size_t)n * KK * CC;
  for (int idx = t; idx < KK * CC; idx += 256) {
    int r = idx / CC, c = idx - r * CC;
    float v = 0.f;
    #pragma unroll
    for (int m = 0; m < KK; ++m) v += adjw[r * KK + m] * xs[m * CC + c];
    outp[idx] = (__bf16)v;
    xout[idx] = (__bf16)xs[idx];
  }
}

// ---- kernel 4: fused WMMA GEMM: out = relu(agg@Wm^T) + x@Wo^T ----
// One 16x32 output tile per wave (two N-tiles share each A fragment ->
// 4 independent WMMA chains, fewer loads per wmma).
// A 16x32 bf16 layout: lanes 0-15 K=0..15, lanes 16-31 K=16..31.
// B 32x16 (KxN): lane = column n, lanes 0-15 K=0..15, lanes 16-31 K=16..31;
// column n of B is row n of W (contiguous).
__device__ __forceinline__ v16bf ld_frag(const __bf16* p, int gap) {
  v8bf a = *(const v8bf*)p;
  v8bf b = *(const v8bf*)(p + gap);
  v16bf r;
  #pragma unroll
  for (int i = 0; i < 8; ++i) { r[i] = a[i]; r[8 + i] = b[i]; }
  return r;
}

__global__ void __launch_bounds__(256)
gemm_kernel(const __bf16* __restrict__ xb,
            const __bf16* __restrict__ agg_b,
            const __bf16* __restrict__ wm_b,
            const __bf16* __restrict__ wo_b,
            float* __restrict__ out) {
  int lane = threadIdx.x & 31;
  int wave = threadIdx.x >> 5;
  int ntp = blockIdx.x;            // 0..15 -> columns [32*ntp, 32*ntp+32)
  int mt  = blockIdx.y * 8 + wave; // 0..1791
  int m0 = mt * 16;
  int n0 = ntp * 32;

  int row_a = m0 + (lane & 15);
  int koffA = (lane < 16) ? 0 : 16;
  int koffB = (lane < 16) ? 0 : 16;
  int cb0 = n0 + (lane & 15);
  int cb1 = cb0 + 16;

  const __bf16* agp = agg_b + (size_t)row_a * CC;
  const __bf16* xp  = xb    + (size_t)row_a * CC;
  const __bf16* wm0 = wm_b  + (size_t)cb0 * CC;
  const __bf16* wm1 = wm_b  + (size_t)cb1 * CC;
  const __bf16* wo0 = wo_b  + (size_t)cb0 * CC;
  const __bf16* wo1 = wo_b  + (size_t)cb1 * CC;

  v8f am0 = {}, am1 = {}, ao0 = {}, ao1 = {};
  for (int k0 = 0; k0 < CC; k0 += 32) {
    v16bf Aagg = ld_frag(agp + k0 + koffA, 8);
    v16bf Ax   = ld_frag(xp  + k0 + koffA, 8);
    v16bf Bm0  = ld_frag(wm0 + k0 + koffB, 8);
    v16bf Bm1  = ld_frag(wm1 + k0 + koffB, 8);
    v16bf Bo0  = ld_frag(wo0 + k0 + koffB, 8);
    v16bf Bo1  = ld_frag(wo1 + k0 + koffB, 8);

    am0 = __builtin_amdgcn_wmma_f32_16x16x32_bf16(false, Aagg, false, Bm0,
                                                  (short)0, am0, false, false);
    am1 = __builtin_amdgcn_wmma_f32_16x16x32_bf16(false, Aagg, false, Bm1,
                                                  (short)0, am1, false, false);
    ao0 = __builtin_amdgcn_wmma_f32_16x16x32_bf16(false, Ax, false, Bo0,
                                                  (short)0, ao0, false, false);
    ao1 = __builtin_amdgcn_wmma_f32_16x16x32_bf16(false, Ax, false, Bo1,
                                                  (short)0, ao1, false, false);
  }
  // C/D layout: VGPR v -> M = v (lanes 0-15) / v+8 (lanes 16-31); N = lane%16
  int rbase = m0 + ((lane < 16) ? 0 : 8);
  int col0 = n0 + (lane & 15);
  #pragma unroll
  for (int v = 0; v < 8; ++v) {
    float* row = out + (size_t)(rbase + v) * OUTC;
    row[col0]      = fmaxf(am0[v], 0.f) + ao0[v];
    row[col0 + 16] = fmaxf(am1[v], 0.f) + ao1[v];
  }
}

extern "C" void kernel_launch(void* const* d_in, const int* in_sizes, int n_in,
                              void* d_out, int out_size, void* d_ws, size_t ws_size,
                              hipStream_t stream) {
  const float* x     = (const float*)d_in[0];
  const float* adj   = (const float*)d_in[1];
  const int*   ei    = (const int*)d_in[2];
  const int*   ej    = (const int*)d_in[3];
  const float* gamma = (const float*)d_in[4];
  const float* beta  = (const float*)d_in[5];
  const float* wdir  = (const float*)d_in[6];
  const float* Wm    = (const float*)d_in[7];
  const float* Wo    = (const float*)d_in[8];
  float* out = (float*)d_out;

  char* ws = (char*)d_ws;
  float*  sum    = (float*)(ws + 0);
  float*  sumsq  = (float*)(ws + 2048);
  float*  u      = (float*)(ws + 4096);
  float*  bconst = (float*)(ws + 6144);
  __bf16* wm_b   = (__bf16*)(ws + WS_WM);
  __bf16* wo_b   = (__bf16*)(ws + WS_WO);
  __bf16* agg_b  = (__bf16*)(ws + WS_AGG);
  __bf16* xb     = (__bf16*)(ws + WS_XB);

  prep_kernel<<<1024, 256, 0, stream>>>(Wm, Wo, wm_b, wo_b, sum, sumsq);
  stats_kernel<<<dim3(2, 32), 256, 0, stream>>>(x, sum, sumsq);
  finalize_kernel<<<1, 256, 0, stream>>>(gamma, beta, wdir, sum, sumsq, u, bconst);
  adj_agg_kernel<<<N_PTS, 256, 0, stream>>>(x, adj, ei, ej, u, bconst, agg_b, xb);
  // 16 N-tile-pairs in x; 1792 M-tiles / 8 waves = 224 blocks in y
  gemm_kernel<<<dim3(16, 224), 256, 0, stream>>>(xb, agg_b, wm_b, wo_b, out);
}
